// SetAbstractD_51170240364930
// MI455X (gfx1250) — matrix operand order repaired and checked by reading, not compiled, not run on hardware
//
#include <hip/hip_runtime.h>
#include <hip/hip_bf16.h>

typedef __attribute__((ext_vector_type(16))) _Float16 v16h;
typedef __attribute__((ext_vector_type(8)))  _Float16 v8h;
typedef __attribute__((ext_vector_type(8)))  float    v8f;
typedef unsigned int u32;
typedef __attribute__((ext_vector_type(4))) u32 u32x4;
typedef __attribute__((ext_vector_type(8))) int i32x8;
typedef __attribute__((ext_vector_type(4))) int i32x4;

#define BB   16
#define NN   8192
#define SS   2048
#define KK   16
#define DIN  64

__device__ inline v16h cat16(v8h lo, v8h hi) {
    return __builtin_shufflevector(lo, hi, 0, 1, 2, 3, 4, 5, 6, 7,
                                           8, 9, 10, 11, 12, 13, 14, 15);
}

// ---------------------------------------------------------------------------
// TDM: 1-D dword copy global -> LDS via tensor descriptor (D#).
// ---------------------------------------------------------------------------
__device__ inline void tdm_copy_dwords(const void* gsrc, u32 lds_byte_addr, u32 ndw) {
#if __has_builtin(__builtin_amdgcn_tensor_load_to_lds)
    unsigned long long ga = (unsigned long long)gsrc;
    u32x4 g0;
    g0[0] = 1u;                                        // count=1, user mode
    g0[1] = lds_byte_addr;                             // lds_addr
    g0[2] = (u32)(ga & 0xFFFFFFFFull);                 // global_addr[31:0]
    g0[3] = (u32)((ga >> 32) & 0x1FFFFFFull) | (2u << 30);  // addr[56:32] | type=2
    i32x8 g1;
    g1[0] = (int)(2u << 16);                           // data_size = 4 bytes
    g1[1] = (int)((ndw & 0xFFFFu) << 16);              // tensor_dim0[15:0]
    g1[2] = (int)(((ndw >> 16) & 0xFFFFu) | (1u << 16)); // dim0[31:16] | tensor_dim1=1
    g1[3] = (int)((ndw & 0xFFFFu) << 16);              // tile_dim0 = ndw
    g1[4] = 0;                                         // tile_dim1/2 unused
    g1[5] = (int)ndw;                                  // tensor_dim0_stride
    g1[6] = 0;
    g1[7] = 0;
    i32x4 z4 = {0, 0, 0, 0};
#if defined(__clang_major__) && __clang_major__ >= 23
    i32x8 z8 = {0, 0, 0, 0, 0, 0, 0, 0};
    __builtin_amdgcn_tensor_load_to_lds(g0, g1, z4, z4, z8, 0);
#else
    __builtin_amdgcn_tensor_load_to_lds(g0, g1, z4, z4, 0);
#endif
    __builtin_amdgcn_s_wait_tensorcnt(0);
#endif
}

// ---------------------------------------------------------------------------
// Pack weights to f16 in per-lane WMMA B-fragment order:
//   frag(ot,kk): element i of lane L is  w[o = ot*16 + (L&15)][c = kk*32 + (i>>3)*16 + (L>>4)*8 + (i&7)]
// wfp: 4 ot x 2 kk,  w1p: 4 ot x 2 kk,  w2p: 8 ot x 2 kk.
// w0x: padded float4[64] = (wx, wy, wz, 0) per output channel.
// ---------------------------------------------------------------------------
__global__ void pack_weights_kernel(const float* __restrict__ w0,
                                    const float* __restrict__ w1,
                                    const float* __restrict__ w2,
                                    _Float16* __restrict__ wfp,
                                    _Float16* __restrict__ w1p,
                                    _Float16* __restrict__ w2p,
                                    float* __restrict__ w0x) {
    int t = threadIdx.x;
    for (int j = t; j < 4096; j += 256) {              // 4 ot * 2 kk * 32 lanes * 16
        int i = j & 15, lane = (j >> 4) & 31, kk = (j >> 9) & 1, ot = j >> 10;
        int c = kk * 32 + ((i >> 3) << 4) + ((lane >> 4) << 3) + (i & 7);
        int o = ot * 16 + (lane & 15);
        wfp[j] = (_Float16)w0[o * 67 + 3 + c];
        w1p[j] = (_Float16)w1[o * 64 + c];
    }
    for (int j = t; j < 8192; j += 256) {              // 8 ot * 2 kk * 32 lanes * 16
        int i = j & 15, lane = (j >> 4) & 31, kk = (j >> 9) & 1, ot = j >> 10;
        int c = kk * 32 + ((i >> 3) << 4) + ((lane >> 4) << 3) + (i & 7);
        int o = ot * 16 + (lane & 15);
        w2p[j] = (_Float16)w2[o * 64 + c];
    }
    for (int i = t; i < 256; i += 256) { }             // (keep thread count uniform)
    for (int i = t; i < 64 * 4; i += 256) {
        int o = i >> 2, c = i & 3;
        w0x[i] = (c < 3) ? w0[o * 67 + c] : 0.0f;
    }
}

// ---------------------------------------------------------------------------
// Furthest point sampling: one workgroup per batch; cloud + min-dist in LDS.
// ---------------------------------------------------------------------------
__global__ __launch_bounds__(1024) void fps_kernel(const float* __restrict__ xyz,
                                                   float* __restrict__ new_xyz,
                                                   int* __restrict__ fps_idx) {
    __shared__ float px[NN], py[NN], pz[NN], dist[NN];
    __shared__ float wbd[32];
    __shared__ int   wbi[32];
    __shared__ int   cur;

    const int b = blockIdx.x;
    const int t = threadIdx.x;
    const float* xb = xyz + (size_t)b * 3 * NN;

    for (int i = t; i < NN; i += 1024) {
        px[i] = xb[i];
        py[i] = xb[NN + i];
        pz[i] = xb[2 * NN + i];
        dist[i] = 1e10f;
    }
    if (t == 0) cur = 0;
    __syncthreads();

    const int lane = t & 31, wave = t >> 5;

    for (int s = 0; s < SS; ++s) {
        const int far = cur;
        const float fx = px[far], fy = py[far], fz = pz[far];
        if (t == 0) {
            fps_idx[b * SS + s] = far;
            new_xyz[(size_t)b * 3 * SS + 0 * SS + s] = fx;
            new_xyz[(size_t)b * 3 * SS + 1 * SS + s] = fy;
            new_xyz[(size_t)b * 3 * SS + 2 * SS + s] = fz;
        }
        float bd = -1.0f; int bi = 0;
        for (int i = t; i < NN; i += 1024) {
            float dx = px[i] - fx, dy = py[i] - fy, dz = pz[i] - fz;
            float d  = dx * dx + dy * dy + dz * dz;
            float dm = fminf(dist[i], d);
            dist[i] = dm;
            if (dm > bd) { bd = dm; bi = i; }
        }
        #pragma unroll
        for (int off = 16; off > 0; off >>= 1) {
            float od = __shfl_xor(bd, off, 32);
            int   oi = __shfl_xor(bi, off, 32);
            if (od > bd || (od == bd && oi < bi)) { bd = od; bi = oi; }
        }
        if (lane == 0) { wbd[wave] = bd; wbi[wave] = bi; }
        __syncthreads();
        if (wave == 0) {
            float d2 = wbd[lane]; int i2 = wbi[lane];
            #pragma unroll
            for (int off = 16; off > 0; off >>= 1) {
                float od = __shfl_xor(d2, off, 32);
                int   oi = __shfl_xor(i2, off, 32);
                if (od > d2 || (od == d2 && oi < i2)) { d2 = od; i2 = oi; }
            }
            if (lane == 0) cur = i2;
        }
        __syncthreads();
    }
}

// ---------------------------------------------------------------------------
// feat[b][n][o] = sum_c points[b][c][n] * w0[o][3+c]  (WMMA 16x16x32 f16).
// One wave per 16-point tile; B fragments are packed v16h (2x b128 each).
// ---------------------------------------------------------------------------
__global__ __launch_bounds__(256) void feat_kernel(const float* __restrict__ points,
                                                   const _Float16* __restrict__ wfp,
                                                   _Float16* __restrict__ feat) {
    const int wid  = (blockIdx.x * 256 + threadIdx.x) >> 5;   // 0..B*(N/16)-1
    const int lane = threadIdx.x & 31;
    const int b    = wid >> 9;
    const int n0   = (wid & 511) << 4;
    const float* pb = points + (size_t)b * DIN * NN;

    const int lrow = lane & 15, khalf = lane >> 4;

    v16h a0, a1;
    #pragma unroll
    for (int i = 0; i < 16; ++i) {
        int c = ((i >> 3) << 4) + (khalf << 3) + (i & 7);
        a0[i] = (_Float16)pb[(size_t)c * NN + n0 + lrow];
        a1[i] = (_Float16)pb[(size_t)(c + 32) * NN + n0 + lrow];
    }
    const v16h* wfv = (const v16h*)wfp;
    #pragma unroll
    for (int ot = 0; ot < 4; ++ot) {
        v16h b0 = wfv[(ot * 2 + 0) * 32 + lane];
        v16h b1 = wfv[(ot * 2 + 1) * 32 + lane];
        v8f acc = {};
        acc = __builtin_amdgcn_wmma_f32_16x16x32_f16(false, a0, false, b0, (short)0, acc, false, false);
        acc = __builtin_amdgcn_wmma_f32_16x16x32_f16(false, a1, false, b1, (short)0, acc, false, false);
        #pragma unroll
        for (int r = 0; r < 8; ++r) {
            int row = r + (khalf << 3);
            feat[((size_t)b * NN + n0 + row) * 64 + ot * 16 + (lane & 15)] = (_Float16)acc[r];
        }
    }
}

// ---------------------------------------------------------------------------
// Brute-force KNN: one query/thread, register top-16, float4 LDS tiles.
// ---------------------------------------------------------------------------
#define KTILE 1024
__global__ __launch_bounds__(128) void knn_kernel(const float* __restrict__ xyz,
                                                  const float* __restrict__ new_xyz,
                                                  int* __restrict__ knn) {
    __shared__ __align__(16) float4 t4[KTILE];
    const int q = blockIdx.x * 128 + threadIdx.x;
    const int b = q / SS, sI = q % SS;
    const float* xb = xyz + (size_t)b * 3 * NN;
    const float qx = new_xyz[(size_t)b * 3 * SS + sI];
    const float qy = new_xyz[(size_t)b * 3 * SS + SS + sI];
    const float qz = new_xyz[(size_t)b * 3 * SS + 2 * SS + sI];

    float bd[KK]; int bi[KK];
    #pragma unroll
    for (int k = 0; k < KK; ++k) { bd[k] = 3.4e38f; bi[k] = 0; }

    for (int t0 = 0; t0 < NN; t0 += KTILE) {
        __syncthreads();
        for (int i = threadIdx.x; i < KTILE; i += 128)
            t4[i] = make_float4(xb[t0 + i], xb[NN + t0 + i], xb[2 * NN + t0 + i], 0.0f);
        __syncthreads();
        for (int j = 0; j < KTILE; ++j) {
            float4 p = t4[j];
            float dx = p.x - qx, dy = p.y - qy, dz = p.z - qz;
            float d  = dx * dx + dy * dy + dz * dz;
            if (d < bd[KK - 1]) {
                float cd = d; int ci = t0 + j;
                #pragma unroll
                for (int k = 0; k < KK; ++k) {
                    if (cd < bd[k]) {
                        float td = bd[k]; int ti = bi[k];
                        bd[k] = cd; bi[k] = ci; cd = td; ci = ti;
                    }
                }
            }
        }
    }
    #pragma unroll
    for (int k = 0; k < KK; ++k) knn[(size_t)q * KK + k] = bi[k];
}

// ---------------------------------------------------------------------------
// Grouped MLP: one wave per group. Weights TDM'd into LDS once per block;
// A fragments assembled from LDS as b128 pairs; fused leaky + max-over-K.
// ---------------------------------------------------------------------------
__global__ __launch_bounds__(256) void group_mlp_kernel(const float* __restrict__ xyz,
                                                        const float* __restrict__ new_xyz,
                                                        const _Float16* __restrict__ feat,
                                                        const int* __restrict__ knn,
                                                        const float4* __restrict__ w0x4,
                                                        const _Float16* __restrict__ w1p,
                                                        const _Float16* __restrict__ w2p,
                                                        float* __restrict__ out) {
    __shared__ __align__(16) _Float16 h0s[8][16][64];
    __shared__ __align__(16) _Float16 h1s[8][16][64];
    __shared__ __align__(16) _Float16 w1s[4096];   // 8KB, packed fragments
    __shared__ __align__(16) _Float16 w2s[8192];   // 16KB, packed fragments

    const int wv = threadIdx.x >> 5, lane = threadIdx.x & 31;
    const int g  = blockIdx.x * 8 + wv;
    const int b  = g / SS, sI = g % SS;
    const int lrow = lane & 15, khalf = lane >> 4, col = lane & 15;

#if __has_builtin(__builtin_amdgcn_tensor_load_to_lds)
    if (wv == 0) {
        tdm_copy_dwords(w1p, (u32)(size_t)(void*)w1s, 4096 * 2 / 4);
        tdm_copy_dwords(w2p, (u32)(size_t)(void*)w2s, 8192 * 2 / 4);
    }
#else
    for (int i = threadIdx.x; i < 4096 / 2; i += 256)
        ((u32*)w1s)[i] = ((const u32*)w1p)[i];
    for (int i = threadIdx.x; i < 8192 / 2; i += 256)
        ((u32*)w2s)[i] = ((const u32*)w2p)[i];
#endif

    // ---- layer 0: h0 = leaky(w0xyz . g_xyz + g_feat) -----------------------
    {
        const int idx = knn[(size_t)g * KK + lrow];
        const float* xb = xyz + (size_t)b * 3 * NN;
        const float gx = xb[idx]          - new_xyz[(size_t)b * 3 * SS + sI];
        const float gy = xb[NN + idx]     - new_xyz[(size_t)b * 3 * SS + SS + sI];
        const float gz = xb[2 * NN + idx] - new_xyz[(size_t)b * 3 * SS + 2 * SS + sI];
        const v8h* frv = (const v8h*)(feat + ((size_t)b * NN + idx) * 64);
        v8h* h0row = (v8h*)&h0s[wv][lrow][khalf * 32];
        const int obase = khalf * 32;
        #pragma unroll
        for (int jj = 0; jj < 4; ++jj) {
            v8h f = frv[khalf * 4 + jj];           // 8 contiguous feature halves (b128)
            v8h hv;
            #pragma unroll
            for (int e = 0; e < 8; ++e) {
                int oo = obase + jj * 8 + e;
                float4 w = w0x4[oo];               // single b128 per row
                float h = w.x * gx + w.y * gy + w.z * gz + (float)f[e];
                h = h >= 0.0f ? h : 0.1f * h;
                hv[e] = (_Float16)h;
            }
            h0row[jj] = hv;                        // ds_store_b128
        }
    }
    __syncthreads();

    // ---- layer 1: h1[16][64] = leaky(h0 @ W1^T), WMMA ----------------------
    const v8h* h0v = (const v8h*)&h0s[wv][lrow][0];      // 8-half (16B) units
    v16h a0 = cat16(h0v[khalf],     h0v[2 + khalf]);     // c: kh*8..+8, 16+kh*8..+8
    v16h a1 = cat16(h0v[4 + khalf], h0v[6 + khalf]);     // +32
    const v8h* w1v = (const v8h*)w1s;
    #pragma unroll
    for (int ot = 0; ot < 4; ++ot) {
        int f0 = ((ot * 2 + 0) * 32 + lane) * 2;
        int f1 = ((ot * 2 + 1) * 32 + lane) * 2;
        v16h b0 = cat16(w1v[f0], w1v[f0 + 1]);
        v16h b1 = cat16(w1v[f1], w1v[f1 + 1]);
        v8f acc = {};
        acc = __builtin_amdgcn_wmma_f32_16x16x32_f16(false, a0, false, b0, (short)0, acc, false, false);
        acc = __builtin_amdgcn_wmma_f32_16x16x32_f16(false, a1, false, b1, (short)0, acc, false, false);
        #pragma unroll
        for (int r = 0; r < 8; ++r) {
            float h = acc[r];
            h = h >= 0.0f ? h : 0.1f * h;
            h1s[wv][r + (khalf << 3)][ot * 16 + col] = (_Float16)h;
        }
    }
    __syncthreads();

    // ---- layer 2: [16][128] = leaky(h1 @ W2^T), WMMA + max over K ----------
    const v8h* h1v = (const v8h*)&h1s[wv][lrow][0];
    a0 = cat16(h1v[khalf],     h1v[2 + khalf]);
    a1 = cat16(h1v[4 + khalf], h1v[6 + khalf]);
    const v8h* w2v = (const v8h*)w2s;
    float* ob = out + (size_t)b * 128 * SS;
    #pragma unroll
    for (int ot = 0; ot < 8; ++ot) {
        int f0 = ((ot * 2 + 0) * 32 + lane) * 2;
        int f1 = ((ot * 2 + 1) * 32 + lane) * 2;
        v16h b0 = cat16(w2v[f0], w2v[f0 + 1]);
        v16h b1 = cat16(w2v[f1], w2v[f1 + 1]);
        v8f acc = {};
        acc = __builtin_amdgcn_wmma_f32_16x16x32_f16(false, a0, false, b0, (short)0, acc, false, false);
        acc = __builtin_amdgcn_wmma_f32_16x16x32_f16(false, a1, false, b1, (short)0, acc, false, false);
        float m = -3.4e38f;
        #pragma unroll
        for (int r = 0; r < 8; ++r) {
            float h = acc[r];
            h = h >= 0.0f ? h : 0.1f * h;
            m = fmaxf(m, h);
        }
        m = fmaxf(m, __shfl_xor(m, 16, 32));
        if (khalf == 0) ob[(size_t)(ot * 16 + col) * SS + sI] = m;
    }
}

// ---------------------------------------------------------------------------
extern "C" void kernel_launch(void* const* d_in, const int* in_sizes, int n_in,
                              void* d_out, int out_size, void* d_ws, size_t ws_size,
                              hipStream_t stream) {
    const float* xyz    = (const float*)d_in[0];
    const float* points = (const float*)d_in[1];
    const float* w0     = (const float*)d_in[2];
    const float* w1     = (const float*)d_in[3];
    const float* w2     = (const float*)d_in[4];

    float* out_all  = (float*)d_out;
    float* new_xyz  = out_all;                                    // [B,3,S]
    float* out_feat = out_all + (size_t)BB * 3 * SS;              // [B,128,S]
    int*   fps_idx  = (int*)(out_all + (size_t)BB * 3 * SS + (size_t)BB * 128 * SS);

    char* ws = (char*)d_ws;
    size_t off = 0;
    _Float16* feat = (_Float16*)(ws + off); off += (size_t)BB * NN * 64 * 2;
    off = (off + 255) & ~(size_t)255;
    _Float16* wfp  = (_Float16*)(ws + off); off += 4096 * 2;
    off = (off + 255) & ~(size_t)255;
    _Float16* w1p  = (_Float16*)(ws + off); off += 4096 * 2;
    off = (off + 255) & ~(size_t)255;
    _Float16* w2p  = (_Float16*)(ws + off); off += 8192 * 2;
    off = (off + 255) & ~(size_t)255;
    float*    w0x  = (float*)(ws + off);    off += 64 * 4 * 4;
    off = (off + 255) & ~(size_t)255;
    int*      knn  = (int*)(ws + off);      off += (size_t)BB * SS * KK * 4;

    pack_weights_kernel<<<1, 256, 0, stream>>>(w0, w1, w2, wfp, w1p, w2p, w0x);
    fps_kernel<<<BB, 1024, 0, stream>>>(xyz, new_xyz, fps_idx);
    feat_kernel<<<(BB * (NN / 16) * 32) / 256, 256, 0, stream>>>(points, wfp, feat);
    knn_kernel<<<(BB * SS) / 128, 128, 0, stream>>>(xyz, new_xyz, knn);
    group_mlp_kernel<<<(BB * SS) / 8, 256, 0, stream>>>(xyz, new_xyz, feat, knn,
                                                        (const float4*)w0x, w1p, w2p, out_feat);
}
